// Generator_36842229465596
// MI455X (gfx1250) — matrix operand order repaired and checked by reading, not compiled, not run on hardware
//
#include <hip/hip_runtime.h>

// ---------------------------------------------------------------------------
// Problem constants (from reference)
// ---------------------------------------------------------------------------
#define BSZ   16384
#define HDIM  256
#define FDIM  128
#define ZDIM  64
#define PREDN 12
#define PREVN 3
#define DDEC  704    // z(64) | h_i(256) | y_prev(128) | h(256)
#define NGATE 1280   // i,f,o,g (1024) | xWxt (256)
#define DINF  768    // h_i | c | y | y_prev
#define DPRI  640    // h_i | c | y_prev

typedef __attribute__((ext_vector_type(16))) __bf16        v16bf;
typedef __attribute__((ext_vector_type(8)))  float          v8f;
typedef __attribute__((ext_vector_type(8)))  unsigned short ush8;

union FragU { v16bf v; ush8 h[2]; };

__device__ __forceinline__ unsigned short f2bf(float f) {
    union { float f; unsigned u; } c; c.f = f;
    unsigned u = c.u;
    unsigned r = u + 0x7fffu + ((u >> 16) & 1u);   // round-to-nearest-even
    return (unsigned short)(r >> 16);
}
__device__ __forceinline__ float sigm(float x) { return 1.0f / (1.0f + __expf(-x)); }

// ---------------------------------------------------------------------------
// Generic bf16 WMMA GEMM:  C[M,N] = A[M,K] * W[K,N] (+bias)(+relu)
//   A  : bf16 row-major, leading dim lda
//   Wt : bf16, W TRANSPOSED, [N,K] row-major (contiguous K for LDS staging)
//   Cf : optional fp32 out, leading dim ldcf
//   Ch : optional bf16 out, leading dim ldch
// Block: 256 thr = 8 waves (4x2), tile 128(M) x 128(N), K step 32.
// Wave tile 32x64 = 2x4 wmma 16x16 tiles. Double-buffered LDS: one barrier
// per K step; next slab's global loads overlap the current slab's WMMAs.
// M must be a multiple of 128, K a multiple of 32 (true at all call sites).
// ---------------------------------------------------------------------------
__global__ __launch_bounds__(256) void gemm_bf16_wmma(
    const unsigned short* __restrict__ A, int lda,
    const unsigned short* __restrict__ Wt,
    const float* __restrict__ bias,
    float* __restrict__ Cf, int ldcf,
    unsigned short* __restrict__ Ch, int ldch,
    int N, int K, int relu)
{
    __shared__ unsigned short As[2][128 * 40];   // 128 rows x 32 k, stride 40
    __shared__ unsigned short Bs[2][128 * 40];   // 128 n    x 32 k, stride 40

    const int tid   = threadIdx.x;
    const int lane  = tid & 31;
    const int w     = tid >> 5;
    const int wm    = w >> 1;          // 0..3
    const int wn    = w & 1;           // 0..1
    const int l15   = lane & 15;
    const int hi16  = (lane >= 16);
    const int mBase = blockIdx.y * 128;
    const int nBase = blockIdx.x * 128;

    const int rb0 = tid >> 2;          // staging row base (stride 64 per it)
    const int c8  = (tid & 3) * 8;     // staging k-offset (halves)

    const v8f zf = {0.f, 0.f, 0.f, 0.f, 0.f, 0.f, 0.f, 0.f};
    v8f acc[2][4];
#pragma unroll
    for (int mi = 0; mi < 2; ++mi)
#pragma unroll
        for (int ni = 0; ni < 4; ++ni) acc[mi][ni] = zf;

    const ush8 zh = {0, 0, 0, 0, 0, 0, 0, 0};
    ush8 ra[2], rb[2];

    // ---- stage slab kt into registers (A: always in range; B: guard N)
    auto load_slab = [&](int kt) {
#pragma unroll
        for (int it = 0; it < 2; ++it) {
            int r = rb0 + it * 64;
            ra[it] = *reinterpret_cast<const ush8*>(
                A + (size_t)(mBase + r) * lda + kt + c8);
            rb[it] = (nBase + r < N)
                ? *reinterpret_cast<const ush8*>(
                      Wt + (size_t)(nBase + r) * K + kt + c8)
                : zh;
        }
    };
    auto store_slab = [&](int p) {
#pragma unroll
        for (int it = 0; it < 2; ++it) {
            int r = rb0 + it * 64;
            *reinterpret_cast<ush8*>(&As[p][r * 40 + c8]) = ra[it];
            *reinterpret_cast<ush8*>(&Bs[p][r * 40 + c8]) = rb[it];
        }
    };
    // NOTE: 128 rows x 32 halves per tile = 4096 halves = 512 x b128;
    // 256 threads x 2 iterations covers it (it stride 64 rows).

    const int nIter = K / 32;
    load_slab(0);
    store_slab(0);

    for (int i = 0; i < nIter; ++i) {
        const int p = i & 1;
        if (i + 1 < nIter) load_slab((i + 1) * 32);  // overlap with WMMAs
        __syncthreads();                              // buf p ready, p^1 free

        // A 16x32 bf16: lanes 0-15 hold K 0..7 & 16..23; lanes 16-31 K 8..15 & 24..31
        FragU af[2];
        const int abase = hi16 ? 8 : 0;
#pragma unroll
        for (int mi = 0; mi < 2; ++mi) {
            int r = wm * 32 + mi * 16 + l15;
            af[mi].h[0] = *reinterpret_cast<const ush8*>(&As[p][r * 40 + abase]);
            af[mi].h[1] = *reinterpret_cast<const ush8*>(&As[p][r * 40 + abase + 16]);
        }
        // B 32x16: lanes 0-15 hold K 0..7 & 16..23; lanes 16-31 K 8..15 & 24..31 (N = lane&15)
        FragU bfr[4];
        const int kb = hi16 ? 8 : 0;
#pragma unroll
        for (int ni = 0; ni < 4; ++ni) {
            int n = wn * 64 + ni * 16 + l15;
            bfr[ni].h[0] = *reinterpret_cast<const ush8*>(&Bs[p][n * 40 + kb]);
            bfr[ni].h[1] = *reinterpret_cast<const ush8*>(&Bs[p][n * 40 + kb + 16]);
        }
#pragma unroll
        for (int mi = 0; mi < 2; ++mi)
#pragma unroll
            for (int ni = 0; ni < 4; ++ni)
                acc[mi][ni] = __builtin_amdgcn_wmma_f32_16x16x32_bf16(
                    false, af[mi].v, false, bfr[ni].v,
                    (short)0, acc[mi][ni], false, false);

        if (i + 1 < nIter) store_slab(p ^ 1);
    }

    // ---- epilogue: C 16x16 f32: row = r + (lane>=16 ? 8:0), col = lane&15
    const int rofs = hi16 ? 8 : 0;
#pragma unroll
    for (int mi = 0; mi < 2; ++mi) {
#pragma unroll
        for (int ni = 0; ni < 4; ++ni) {
            int col = nBase + wn * 64 + ni * 16 + l15;
            if (col < N) {
                float bv = bias ? bias[col] : 0.f;
#pragma unroll
                for (int r = 0; r < 8; ++r) {
                    int row = mBase + wm * 32 + mi * 16 + r + rofs;
                    float v = acc[mi][ni][r] + bv;
                    if (relu) v = v > 0.f ? v : 0.f;
                    if (Cf) Cf[(size_t)row * ldcf + col] = v;
                    if (Ch) Ch[(size_t)row * ldch + col] = f2bf(v);
                }
            }
        }
    }
}

// ---------------------------------------------------------------------------
// Weight prep kernels (fp32 -> bf16 transposed)
// ---------------------------------------------------------------------------
__global__ void k_tr_bf16(unsigned short* __restrict__ dst,
                          const float* __restrict__ src,
                          int K, int N, int dstld, int rowoff)
{
    int idx = blockIdx.x * blockDim.x + threadIdx.x;
    if (idx >= K * N) return;
    int k = idx / N, n = idx % N;
    dst[(size_t)(n + rowoff) * dstld + k] = f2bf(src[(size_t)k * N + n]);
}

// Combined gate weight: Wg[K=704, N=1280], stored transposed [1280,704]
__global__ void k_build_wg(unsigned short* __restrict__ dst,
                           const float* __restrict__ Wx,   // [448,1024]
                           const float* __restrict__ Wxt,  // [448,256]
                           const float* __restrict__ Wh)   // [256,1024]
{
    int idx = blockIdx.x * blockDim.x + threadIdx.x;
    if (idx >= NGATE * DDEC) return;
    int n = idx / DDEC, k = idx % DDEC;
    float v;
    if (k < 448) v = (n < 1024) ? Wx[(size_t)k * 1024 + n] : Wxt[(size_t)k * 256 + (n - 1024)];
    else         v = (n < 1024) ? Wh[(size_t)(k - 448) * 1024 + n] : 0.f;
    dst[(size_t)n * DDEC + k] = f2bf(v);
}

__global__ void k_cat_bias(float* __restrict__ dst,
                           const float* __restrict__ a, const float* __restrict__ b)
{
    int i = threadIdx.x;          // launched with 128 threads
    dst[i] = (i < 64) ? a[i] : b[i - 64];
}

__global__ void k_zero_u32(unsigned* __restrict__ p, long long n)
{
    long long i = (long long)blockIdx.x * blockDim.x + threadIdx.x;
    if (i < n) p[i] = 0u;
}

// h_i (fp32) -> bf16 into the three concat buffers' h_i slots
__global__ void k_fill_hi(const float* __restrict__ h_i,
                          unsigned short* __restrict__ adec,
                          unsigned short* __restrict__ ainf,
                          unsigned short* __restrict__ apri)
{
    int idx = blockIdx.x * blockDim.x + threadIdx.x;
    if (idx >= BSZ * HDIM) return;
    int b = idx >> 8, n = idx & 255;
    unsigned short v = f2bf(h_i[idx]);
    adec[(size_t)b * DDEC + 64 + n]  = v;
    ainf[(size_t)b * DINF + n]       = v;
    apri[(size_t)b * DPRI + n]       = v;
}

// Propagate previous y (Ainf y-slot) into all y_prev slots
__global__ void k_copy_yprev(const unsigned short* __restrict__ ainf,
                             unsigned short* __restrict__ adec,
                             unsigned short* __restrict__ ainf_w,
                             unsigned short* __restrict__ apri)
{
    int idx = blockIdx.x * blockDim.x + threadIdx.x;
    if (idx >= BSZ * FDIM) return;
    int b = idx >> 7, f = idx & 127;
    unsigned short v = ainf[(size_t)b * DINF + 512 + f];
    adec[(size_t)b * DDEC + 320 + f]   = v;
    ainf_w[(size_t)b * DINF + 640 + f] = v;
    apri[(size_t)b * DPRI + 512 + f]   = v;
}

// TLSTM-1 pointwise: gates -> c_new, h_new (+ bf16 copies into concat buffers)
__global__ void k_lstm_pw(const float* __restrict__ G,      // [B,1280] raw accum
                          float* __restrict__ c,            // [B,256] in/out
                          unsigned short* __restrict__ hb,  // [B,256] bf16
                          unsigned short* __restrict__ adec,
                          unsigned short* __restrict__ ainf,
                          unsigned short* __restrict__ apri,
                          const float* __restrict__ bg,
                          const float* __restrict__ btt,
                          const float* __restrict__ wtt,
                          const float* __restrict__ wto,
                          const float* __restrict__ input_t,
                          int j)
{
    int idx = blockIdx.x * blockDim.x + threadIdx.x;
    if (idx >= BSZ * HDIM) return;
    int b = idx >> 8, n = idx & 255;
    float t = input_t[(size_t)b * 16 + PREVN + j];
    const float* g = G + (size_t)b * NGATE;
    float gi = g[n]        + bg[n];
    float gf = g[n + 256]  + bg[n + 256];
    float go = g[n + 512]  + bg[n + 512] + t * wto[n];
    float gc = g[n + 768]  + bg[n + 768];
    float xt = g[n + 1024];
    float T  = sigm(xt + sigm(t * wtt[n]) + btt[n]);
    float cn = sigm(gf) * c[idx] + sigm(gi) * T * tanhf(gc);
    float hn = sigm(go) * tanhf(cn);
    c[idx] = cn;
    unsigned short cn16 = f2bf(cn);
    hb[idx] = f2bf(hn);
    adec[(size_t)b * DDEC + 448 + n] = f2bf(hn);
    ainf[(size_t)b * DINF + 256 + n] = cn16;
    apri[(size_t)b * DPRI + 256 + n] = cn16;
}

// Reparameterization: mlv[B,128] = [mean|logvar] -> z = mean + n*sqrt(exp(lv))
__global__ void k_sample(const float* __restrict__ mlv,
                         const float* __restrict__ noise,   // [B,64]
                         float* __restrict__ mean_out,      // nullable, ld 768
                         float* __restrict__ logv_out,      // nullable, ld 768
                         float* __restrict__ z_out,         // ld 768
                         unsigned short* __restrict__ zslot, // nullable
                         int zld)
{
    int idx = blockIdx.x * blockDim.x + threadIdx.x;
    if (idx >= BSZ * ZDIM) return;
    int b = idx >> 6, zn = idx & 63;
    float m  = mlv[(size_t)b * 128 + zn];
    float lv = mlv[(size_t)b * 128 + 64 + zn];
    float zv = m + noise[idx] * sqrtf(__expf(lv));
    size_t o = (size_t)b * (PREDN * ZDIM) + zn;
    if (mean_out) mean_out[o] = m;
    if (logv_out) logv_out[o] = lv;
    z_out[o] = zv;
    if (zslot) zslot[(size_t)b * zld + zn] = f2bf(zv);
}

// ---------------------------------------------------------------------------
extern "C" void kernel_launch(void* const* d_in, const int* in_sizes, int n_in,
                              void* d_out, int out_size, void* d_ws, size_t ws_size,
                              hipStream_t stream)
{
    (void)in_sizes; (void)n_in; (void)out_size; (void)ws_size;
    const float* h_i     = (const float*)d_in[0];
    const float* input_t = (const float*)d_in[1];
    const float* n_inf   = (const float*)d_in[2];
    const float* n_pri   = (const float*)d_in[3];
    const float* Wx  = (const float*)d_in[4];
    const float* Wh  = (const float*)d_in[5];
    const float* bg  = (const float*)d_in[6];
    const float* Wxt = (const float*)d_in[7];
    const float* wtt = (const float*)d_in[8];
    const float* btt = (const float*)d_in[9];
    const float* wto = (const float*)d_in[10];
    const float* W1  = (const float*)d_in[11]; const float* b1 = (const float*)d_in[12];
    const float* W2  = (const float*)d_in[13]; const float* b2 = (const float*)d_in[14];
    const float* W3  = (const float*)d_in[15]; const float* b3 = (const float*)d_in[16];
    const float* W4  = (const float*)d_in[17]; const float* b4 = (const float*)d_in[18];
    const float* W5  = (const float*)d_in[19]; const float* b5 = (const float*)d_in[20];
    const float* W6  = (const float*)d_in[21]; const float* b6 = (const float*)d_in[22];
    const float* W7  = (const float*)d_in[23]; const float* b7 = (const float*)d_in[24];
    const float* W8  = (const float*)d_in[25]; const float* b8 = (const float*)d_in[26];
    const float* W9  = (const float*)d_in[27]; const float* b9 = (const float*)d_in[28];

    // output slices (return order: ys, means, logvs, z_infs, z_pris)
    float* ys     = (float*)d_out;
    float* means  = ys     + (size_t)BSZ * PREDN * FDIM;
    float* logvs  = means  + (size_t)BSZ * PREDN * ZDIM;
    float* z_infs = logvs  + (size_t)BSZ * PREDN * ZDIM;
    float* z_pris = z_infs + (size_t)BSZ * PREDN * ZDIM;

    // workspace carve-out
    char* w = (char*)d_ws;
    size_t off = 0;
    auto carve = [&](size_t bytes) -> void* {
        void* p = w + off;
        off = (off + bytes + 255) & ~(size_t)255;
        return p;
    };
    unsigned short* WgT  = (unsigned short*)carve((size_t)NGATE * DDEC * 2);
    unsigned short* W1T  = (unsigned short*)carve((size_t)128 * 256 * 2);
    unsigned short* W2T  = (unsigned short*)carve((size_t)128 * 128 * 2);
    unsigned short* W3T  = (unsigned short*)carve((size_t)128 * 128 * 2);
    unsigned short* W4T  = (unsigned short*)carve((size_t)64 * DINF * 2);
    unsigned short* W56T = (unsigned short*)carve((size_t)128 * 64 * 2);
    unsigned short* W7T  = (unsigned short*)carve((size_t)64 * DPRI * 2);
    unsigned short* W89T = (unsigned short*)carve((size_t)128 * 64 * 2);
    float*          b56  = (float*)carve(128 * 4);
    float*          b89  = (float*)carve(128 * 4);
    unsigned short* Adec = (unsigned short*)carve((size_t)BSZ * DDEC * 2);
    unsigned short* Ainf = (unsigned short*)carve((size_t)BSZ * DINF * 2);
    unsigned short* Apri = (unsigned short*)carve((size_t)BSZ * DPRI * 2);
    float*          cbuf = (float*)carve((size_t)BSZ * HDIM * 4);
    unsigned short* hbuf = (unsigned short*)carve((size_t)BSZ * HDIM * 2);
    unsigned short* y1   = (unsigned short*)carve((size_t)BSZ * FDIM * 2);
    unsigned short* y2   = (unsigned short*)carve((size_t)BSZ * FDIM * 2);
    unsigned short* hz   = (unsigned short*)carve((size_t)BSZ * ZDIM * 2);
    float*          mlv  = (float*)carve((size_t)BSZ * 128 * 4);
    float*          G    = (float*)carve((size_t)BSZ * NGATE * 4);

    const int TPB = 256;
    auto blks = [](long long n) { return (unsigned)((n + 255) / 256); };

    // ---- one-time weight prep (deterministic every call)
    k_build_wg<<<blks((long long)NGATE * DDEC), TPB, 0, stream>>>(WgT, Wx, Wxt, Wh);
    k_tr_bf16<<<blks(256 * 128), TPB, 0, stream>>>(W1T, W1, 256, 128, 256, 0);
    k_tr_bf16<<<blks(128 * 128), TPB, 0, stream>>>(W2T, W2, 128, 128, 128, 0);
    k_tr_bf16<<<blks(128 * 128), TPB, 0, stream>>>(W3T, W3, 128, 128, 128, 0);
    k_tr_bf16<<<blks(DINF * 64), TPB, 0, stream>>>(W4T, W4, DINF, 64, DINF, 0);
    k_tr_bf16<<<blks(64 * 64), TPB, 0, stream>>>(W56T, W5, 64, 64, 64, 0);
    k_tr_bf16<<<blks(64 * 64), TPB, 0, stream>>>(W56T, W6, 64, 64, 64, 64);
    k_tr_bf16<<<blks(DPRI * 64), TPB, 0, stream>>>(W7T, W7, DPRI, 64, DPRI, 0);
    k_tr_bf16<<<blks(64 * 64), TPB, 0, stream>>>(W89T, W8, 64, 64, 64, 0);
    k_tr_bf16<<<blks(64 * 64), TPB, 0, stream>>>(W89T, W9, 64, 64, 64, 64);
    k_cat_bias<<<1, 128, 0, stream>>>(b56, b5, b6);
    k_cat_bias<<<1, 128, 0, stream>>>(b89, b8, b9);

    // ---- zero-init recurrent state / concat buffers, then fill h_i slots
    k_zero_u32<<<blks((long long)BSZ * DDEC / 2), TPB, 0, stream>>>((unsigned*)Adec, (long long)BSZ * DDEC / 2);
    k_zero_u32<<<blks((long long)BSZ * DINF / 2), TPB, 0, stream>>>((unsigned*)Ainf, (long long)BSZ * DINF / 2);
    k_zero_u32<<<blks((long long)BSZ * DPRI / 2), TPB, 0, stream>>>((unsigned*)Apri, (long long)BSZ * DPRI / 2);
    k_zero_u32<<<blks((long long)BSZ * HDIM), TPB, 0, stream>>>((unsigned*)cbuf, (long long)BSZ * HDIM);
    k_fill_hi<<<blks((long long)BSZ * HDIM), TPB, 0, stream>>>(h_i, Adec, Ainf, Apri);

    const unsigned gM = BSZ / 128;  // 128 M-tiles

    for (int j = 0; j < PREDN; ++j) {
        // y_{j-1} -> y_prev slots (reads Ainf y-slot, zero at j==0)
        k_copy_yprev<<<blks((long long)BSZ * FDIM), TPB, 0, stream>>>(Ainf, Adec, Ainf, Apri);

        // gates_ext = [z|h_i|y_prev|h] @ [Wx|Wxt ; Wh|0]   (K=704, N=1280)
        gemm_bf16_wmma<<<dim3(NGATE / 128, gM), TPB, 0, stream>>>(
            Adec, DDEC, WgT, nullptr, G, NGATE, nullptr, 0, NGATE, DDEC, 0);

        // TLSTM pointwise -> c, h (bf16 copies into Adec/Ainf/Apri)
        k_lstm_pw<<<blks((long long)BSZ * HDIM), TPB, 0, stream>>>(
            G, cbuf, hbuf, Adec, Ainf, Apri, bg, btt, wtt, wto, input_t, j);

        // dense1..3 (relu); dense3 writes fp32 to d_out[ys] and bf16 into Ainf y-slot
        gemm_bf16_wmma<<<dim3(1, gM), TPB, 0, stream>>>(
            hbuf, HDIM, W1T, b1, nullptr, 0, y1, FDIM, FDIM, HDIM, 1);
        gemm_bf16_wmma<<<dim3(1, gM), TPB, 0, stream>>>(
            y1, FDIM, W2T, b2, nullptr, 0, y2, FDIM, FDIM, FDIM, 1);
        gemm_bf16_wmma<<<dim3(1, gM), TPB, 0, stream>>>(
            y2, FDIM, W3T, b3, ys + (size_t)j * FDIM, PREDN * FDIM,
            Ainf + 512, DINF, FDIM, FDIM, 1);

        // inference: hz = relu(Ainf @ W4 + b4); [mean|logv] = relu(hz @ [W5|W6] + b)
        gemm_bf16_wmma<<<dim3(1, gM), TPB, 0, stream>>>(
            Ainf, DINF, W4T, b4, nullptr, 0, hz, ZDIM, ZDIM, DINF, 1);
        gemm_bf16_wmma<<<dim3(1, gM), TPB, 0, stream>>>(
            hz, ZDIM, W56T, b56, mlv, 128, nullptr, 0, 128, ZDIM, 1);
        k_sample<<<blks((long long)BSZ * ZDIM), TPB, 0, stream>>>(
            mlv, n_inf + (size_t)j * BSZ * ZDIM,
            means + (size_t)j * ZDIM, logvs + (size_t)j * ZDIM,
            z_infs + (size_t)j * ZDIM, Adec, DDEC);

        // prior: hz_ = relu(Apri @ W7 + b7); z_pri sampled, outputs only
        gemm_bf16_wmma<<<dim3(1, gM), TPB, 0, stream>>>(
            Apri, DPRI, W7T, b7, nullptr, 0, hz, ZDIM, ZDIM, DPRI, 1);
        gemm_bf16_wmma<<<dim3(1, gM), TPB, 0, stream>>>(
            hz, ZDIM, W89T, b89, mlv, 128, nullptr, 0, 128, ZDIM, 1);
        k_sample<<<blks((long long)BSZ * ZDIM), TPB, 0, stream>>>(
            mlv, n_pri + (size_t)j * BSZ * ZDIM,
            nullptr, nullptr, z_pris + (size_t)j * ZDIM, nullptr, 0);
    }
}
